// GroupQueryAttention_63848983823025
// MI455X (gfx1250) — matrix operand order repaired and checked by reading, not compiled, not run on hardware
//
#include <hip/hip_runtime.h>
#include <stdint.h>

#define BATCH 2
#define SEQ   2048
#define HIDN  2048
#define NH    16
#define HD    128   // head dim (G*D with G=1)

// ---------------- CDNA5 WMMA types ----------------
typedef __attribute__((ext_vector_type(16))) __bf16 v16bf;
typedef __attribute__((ext_vector_type(8)))  float  v8f;
typedef uint32_t __attribute__((may_alias)) u32a;
typedef uint4    __attribute__((may_alias)) u128a;
typedef int v4i_gcc __attribute__((vector_size(16)));  // matches builtin proto

union AFrag { v16bf v; uint32_t u[8]; };

// Async global->LDS DMA (ASYNCcnt path), guarded for toolchain portability.
#if defined(__gfx1250__) &&                                              \
    __has_builtin(__builtin_amdgcn_global_load_async_to_lds_b128) &&     \
    __has_builtin(__builtin_amdgcn_s_wait_asynccnt)
#define USE_ASYNC_LDS 1
#else
#define USE_ASYNC_LDS 0
#endif

#if USE_ASYNC_LDS
__device__ __forceinline__ void async_copy_b128(const unsigned short* g,
                                                unsigned short* l) {
  // emits global_load_async_to_lds_b128 (memory -> LDS, no VGPR round-trip)
  __builtin_amdgcn_global_load_async_to_lds_b128(
      (__attribute__((address_space(1))) v4i_gcc*)g,
      (__attribute__((address_space(3))) v4i_gcc*)l, 0, 0);
}
#endif

__device__ __forceinline__ unsigned short f2bf(float f) {
  union { float f; uint32_t u; } x; x.f = f;
  uint32_t r = x.u + 0x7FFFu + ((x.u >> 16) & 1u);   // round-to-nearest-even
  return (unsigned short)(r >> 16);
}

// K-offsets inside a fragment (ISA 7.12.2, 16-bit data, wave32)
// A 16x32:  VGPR v, lane-half h -> k = 2v (+8 if v>=4) + 8h, pair {k,k+1}
__device__ __forceinline__ int kA(int v, int half) {
  return 2 * v + (v >= 4 ? 8 : 0) + 8 * half;
}
// B 32x16 (column-striped): VGPR v, lane-half h -> k = 2v (+8 if v>=4) + 8h, pair {k,k+1}
__device__ __forceinline__ int kB(int v, int half) {
  return 2 * v + (v >= 4 ? 8 : 0) + 8 * half;
}

// ---------------- fp32 -> bf16 conversion ----------------
__global__ void cvt_f32_to_bf16(const float* __restrict__ src,
                                unsigned short* __restrict__ dst, size_t n) {
  size_t i = (size_t)blockIdx.x * blockDim.x + threadIdx.x;
  size_t stride = (size_t)gridDim.x * blockDim.x;
  for (; i < n; i += stride) dst[i] = f2bf(src[i]);
}

// ---------------- generic C = A * W^T + bias  (bf16 in, f32 acc) ----------
// A:[M,K] bf16 row-major, W:[N,K] bf16 row-major, out:[M,N] (bf16 or f32).
// Block tile 128x128, BK=32. 8 waves: wave = (wm in 0..1)*64 rows x (wn in 0..3)*32 cols.
// Async build: double-buffered LDS, next slab DMA'd while WMMAs run.
template <bool BF16OUT>
__global__ __launch_bounds__(256)
void gemm_bias_wmma(const unsigned short* __restrict__ A,
                    const unsigned short* __restrict__ W,
                    const float* __restrict__ bias,
                    void* __restrict__ out,
                    int M, int N, int K) {
  constexpr int LDT = 40;        // 32 + 8 pad (80B row, 16B aligned)
  constexpr int TSZ = 128 * LDT; // one tile buffer (bf16 elems)
#if USE_ASYNC_LDS
  __shared__ alignas(16) unsigned short As[2 * TSZ];
  __shared__ alignas(16) unsigned short Ws[2 * TSZ];
#else
  __shared__ alignas(16) unsigned short As[TSZ];
  __shared__ alignas(16) unsigned short Ws[TSZ];
#endif

  const int tid  = threadIdx.x;
  const int lane = tid & 31;
  const int wave = tid >> 5;
  const int wm   = wave >> 2;
  const int wn   = wave & 3;
  const int l16  = lane & 15;
  const int half = lane >> 4;
  const int m0   = blockIdx.y * 128;
  const int n0   = blockIdx.x * 128;

  v8f acc[4][2] = {};

  auto compute_tile = [&](const unsigned short* Asb, const unsigned short* Wsb) {
    AFrag af[4], bf[2];
#pragma unroll
    for (int mt = 0; mt < 4; ++mt) {
      const unsigned short* p = Asb + (wm * 64 + mt * 16 + l16) * LDT;
#pragma unroll
      for (int v = 0; v < 8; ++v) af[mt].u[v] = *(const u32a*)(p + kA(v, half));
    }
#pragma unroll
    for (int nt = 0; nt < 2; ++nt) {
      // B[k][n] = W[n][k] -> read row n of staged W
      const unsigned short* p = Wsb + (wn * 32 + nt * 16 + l16) * LDT;
#pragma unroll
      for (int v = 0; v < 8; ++v) bf[nt].u[v] = *(const u32a*)(p + kB(v, half));
    }
#pragma unroll
    for (int mt = 0; mt < 4; ++mt)
#pragma unroll
      for (int nt = 0; nt < 2; ++nt)
        acc[mt][nt] = __builtin_amdgcn_wmma_f32_16x16x32_bf16(
            false, af[mt].v, false, bf[nt].v, (short)0, acc[mt][nt], false, false);
  };

#if USE_ASYNC_LDS
  // prologue: DMA first 128x32 slabs into buffer 0
  for (int i = tid; i < 512; i += 256) {
    int r = i >> 2, c = (i & 3) * 8;
    async_copy_b128(A + (size_t)(m0 + r) * K + c, As + r * LDT + c);
    async_copy_b128(W + (size_t)(n0 + r) * K + c, Ws + r * LDT + c);
  }
  __builtin_amdgcn_s_wait_asynccnt(0);
  __syncthreads();
  int buf = 0;
  for (int k0 = 0; k0 < K; k0 += 32) {
    const int nbuf = buf ^ 1;
    if (k0 + 32 < K) {  // DMA next slab while this one is consumed
      for (int i = tid; i < 512; i += 256) {
        int r = i >> 2, c = (i & 3) * 8;
        async_copy_b128(A + (size_t)(m0 + r) * K + k0 + 32 + c,
                        As + nbuf * TSZ + r * LDT + c);
        async_copy_b128(W + (size_t)(n0 + r) * K + k0 + 32 + c,
                        Ws + nbuf * TSZ + r * LDT + c);
      }
    }
    compute_tile(As + buf * TSZ, Ws + buf * TSZ);
    __builtin_amdgcn_s_wait_asynccnt(0);
    __syncthreads();
    buf = nbuf;
  }
#else
  for (int k0 = 0; k0 < K; k0 += 32) {
    for (int i = tid; i < 512; i += 256) {
      int r = i >> 2, c = (i & 3) * 8;
      *(u128a*)(As + r * LDT + c) =
          *(const u128a*)(A + (size_t)(m0 + r) * K + k0 + c);
      *(u128a*)(Ws + r * LDT + c) =
          *(const u128a*)(W + (size_t)(n0 + r) * K + k0 + c);
    }
    if (k0 + 32 < K) {  // global_prefetch_b8 for the next K-slab
      __builtin_prefetch(A + (size_t)(m0 + (tid >> 1)) * K + k0 + 32, 0, 3);
      __builtin_prefetch(W + (size_t)(n0 + (tid >> 1)) * K + k0 + 32, 0, 3);
    }
    __syncthreads();
    compute_tile(As, Ws);
    __syncthreads();
  }
#endif

  for (int nt = 0; nt < 2; ++nt) {
    const int n   = n0 + wn * 32 + nt * 16 + l16;
    const float bv = bias[n];
    for (int mt = 0; mt < 4; ++mt)
      for (int r = 0; r < 8; ++r) {
        const int m = m0 + wm * 64 + mt * 16 + r + 8 * half;  // C-layout row
        const float val = acc[mt][nt][r] + bv;
        if (BF16OUT)
          ((unsigned short*)out)[(size_t)m * N + n] = f2bf(val);
        else
          ((float*)out)[(size_t)m * N + n] = val;
      }
  }
}

// ---------------- flash GQA attention (bf16 WMMA, online softmax) ---------
// grid: (SEQ/128 query blocks, BATCH*NH). 256 threads = 8 waves,
// wave w owns query rows q0 + 16w .. +15. KV blocks of 128 keys.
__global__ __launch_bounds__(256)
void gqa_flash_wmma(const unsigned short* __restrict__ Qb,  // [B,S,HIDN]
                    const unsigned short* __restrict__ Kb,  // [B,S,HD]
                    const unsigned short* __restrict__ Vb,  // [B,S,HD]
                    const float* __restrict__ pmask,        // [B,S]
                    unsigned short* __restrict__ Ob)        // [B,S,HIDN]
{
  constexpr int LT = 136;  // padded row stride (bf16 elems), 272B, 16B aligned
  extern __shared__ __align__(16) unsigned short smem[];
  unsigned short* Ks = smem;            // [128 key][LT]  (row-major copy of K)
  unsigned short* Vt = Ks + 128 * LT;   // [128 d][LT]    (V transposed)
  unsigned short* Pw = Vt + 128 * LT;   // 8 waves x [16 q][LT] softmax tile

  const int tid  = threadIdx.x;
  const int lane = tid & 31;
  const int wave = tid >> 5;
  const int l16  = lane & 15;
  const int half = lane >> 4;
  const int qb   = blockIdx.x;
  const int b    = blockIdx.y / NH;
  const int h    = blockIdx.y % NH;
  const int q0   = qb * 128;
  const float scl = 0.08838834764831845f;  // 1/sqrt(128)

  // Q A-fragments for this wave's 16 rows; contraction dim = HD = 128 -> 4 slabs
  AFrag qf[4];
  {
    const unsigned short* p =
        Qb + (size_t)(b * SEQ + q0 + wave * 16 + l16) * HIDN + h * HD;
    for (int ks = 0; ks < 4; ++ks)
      for (int v = 0; v < 8; ++v)
        qf[ks].u[v] = *(const u32a*)(p + ks * 32 + kA(v, half));
  }

  v8f accO[8] = {};
  float rm[8], rl[8];
  for (int r = 0; r < 8; ++r) { rm[r] = -__builtin_inff(); rl[r] = 0.f; }
  unsigned short* Pme = Pw + wave * 16 * LT;

  for (int kb = 0; kb <= qb; ++kb) {      // fully-masked blocks skipped
    const int kv0 = kb * 128;
    __syncthreads();
#if USE_ASYNC_LDS
    // K tile via async DMA; V transpose runs in the DMA shadow
    for (int i = tid; i < 2048; i += 256) {
      int r = i >> 4, c = (i & 15) * 8;
      async_copy_b128(Kb + (size_t)(b * SEQ + kv0 + r) * HD + c,
                      Ks + r * LT + c);
    }
#else
    for (int i = tid; i < 2048; i += 256) {
      int r = i >> 4, c = (i & 15) * 8;
      *(u128a*)(Ks + r * LT + c) =
          *(const u128a*)(Kb + (size_t)(b * SEQ + kv0 + r) * HD + c);
    }
#endif
    // stage V transposed: Vt[d][key]
    for (int i = tid; i < 128 * 128; i += 256) {
      int d = i & 127, key = i >> 7;
      Vt[d * LT + key] = Vb[(size_t)(b * SEQ + kv0 + key) * HD + d];
    }
#if USE_ASYNC_LDS
    __builtin_amdgcn_s_wait_asynccnt(0);
#endif
    __syncthreads();

    // scores = Q K^T : B[k=d][n=key] = K[key][d] -> row `key` of Ks
    v8f sc[8] = {};
    for (int nt = 0; nt < 8; ++nt) {
      const unsigned short* kp = Ks + (nt * 16 + l16) * LT;
      for (int ks = 0; ks < 4; ++ks) {
        AFrag bfrag;
        for (int v = 0; v < 8; ++v)
          bfrag.u[v] = *(const u32a*)(kp + ks * 32 + kB(v, half));
        sc[nt] = __builtin_amdgcn_wmma_f32_16x16x32_bf16(
            false, qf[ks].v, false, bfrag.v, (short)0, sc[nt], false, false);
      }
    }

    // scale + causal & padding masks (matches ref: s/sqrt(D) + mask*(-1e9))
    const bool diag = (kb == qb);
    for (int nt = 0; nt < 8; ++nt) {
      const int key = kv0 + nt * 16 + l16;
      const float pm = pmask[b * SEQ + key] * -1e9f;
      for (int r = 0; r < 8; ++r) {
        const int q = q0 + wave * 16 + r + 8 * half;
        float s = sc[nt][r] * scl + pm;
        if (diag && key > q) s += -1e9f;
        sc[nt][r] = s;
      }
    }

    // online softmax; row r+8*half lives in one 16-lane half-wave
    float tmax[8], tsum[8];
    for (int r = 0; r < 8; ++r) {
      float m = sc[0][r];
      for (int nt = 1; nt < 8; ++nt) m = fmaxf(m, sc[nt][r]);
      for (int off = 1; off < 16; off <<= 1)
        m = fmaxf(m, __shfl_xor(m, off, 16));
      tmax[r] = m;
    }
    float nm[8], rs[8];
    for (int r = 0; r < 8; ++r) {
      nm[r] = fmaxf(rm[r], tmax[r]);
      rs[r] = __expf(rm[r] - nm[r]);
      rm[r] = nm[r];
      tsum[r] = 0.f;
    }
    // exponentiate, row sums, spill P (bf16) to wave-private LDS
    for (int nt = 0; nt < 8; ++nt)
      for (int r = 0; r < 8; ++r) {
        const float pv = __expf(sc[nt][r] - nm[r]);
        tsum[r] += pv;
        Pme[(r + 8 * half) * LT + nt * 16 + l16] = f2bf(pv);
      }
    for (int r = 0; r < 8; ++r) {
      float s = tsum[r];
      for (int off = 1; off < 16; off <<= 1) s += __shfl_xor(s, off, 16);
      rl[r] = rl[r] * rs[r] + s;
    }
    for (int nt = 0; nt < 8; ++nt)
      for (int r = 0; r < 8; ++r) accO[nt][r] *= rs[r];

    // O += P V : A = P [16 x 128keys]; B[k=key][n=d] = V[key][d] = Vt[d][key]
    AFrag pf[4];
    {
      const unsigned short* pp = Pme + l16 * LT;
      for (int ks = 0; ks < 4; ++ks)
        for (int v = 0; v < 8; ++v)
          pf[ks].u[v] = *(const u32a*)(pp + ks * 32 + kA(v, half));
    }
    for (int nt = 0; nt < 8; ++nt) {
      const unsigned short* vp = Vt + (nt * 16 + l16) * LT;
      for (int ks = 0; ks < 4; ++ks) {
        AFrag bfrag;
        for (int v = 0; v < 8; ++v)
          bfrag.u[v] = *(const u32a*)(vp + ks * 32 + kB(v, half));
        accO[nt] = __builtin_amdgcn_wmma_f32_16x16x32_bf16(
            false, pf[ks].v, false, bfrag.v, (short)0, accO[nt], false, false);
      }
    }
  }

  // normalize, write context as bf16 into [B,S,H*D] layout
  for (int r = 0; r < 8; ++r) rl[r] = 1.0f / rl[r];
  for (int nt = 0; nt < 8; ++nt) {
    const int d = nt * 16 + l16;
    for (int r = 0; r < 8; ++r) {
      const int q = q0 + wave * 16 + r + 8 * half;
      Ob[(size_t)(b * SEQ + q) * HIDN + h * HD + d] = f2bf(accO[nt][r] * rl[r]);
    }
  }
}

// ---------------- launcher ----------------
extern "C" void kernel_launch(void* const* d_in, const int* in_sizes, int n_in,
                              void* d_out, int out_size, void* d_ws, size_t ws_size,
                              hipStream_t stream) {
  (void)in_sizes; (void)n_in; (void)out_size; (void)ws_size;
  const float* hs    = (const float*)d_in[0];
  // d_in[1] = causal_mask (triu k=1) -> reproduced analytically in-kernel
  const float* pmask = (const float*)d_in[2];
  const float* Wq = (const float*)d_in[3];
  const float* bq = (const float*)d_in[4];
  const float* Wk = (const float*)d_in[5];
  const float* bk = (const float*)d_in[6];
  const float* Wv = (const float*)d_in[7];
  const float* bv = (const float*)d_in[8];
  const float* Wo = (const float*)d_in[9];
  const float* bo = (const float*)d_in[10];

  const size_t nHS = (size_t)BATCH * SEQ * HIDN;  // 8,388,608
  const size_t nWQ = (size_t)HIDN * HIDN;         // 4,194,304
  const size_t nWK = (size_t)HD * HIDN;           //   262,144
  const size_t nKV = (size_t)BATCH * SEQ * HD;    //   524,288

  unsigned short* p   = (unsigned short*)d_ws;    // ~67 MiB of bf16 scratch
  unsigned short* hsb = p; p += nHS;
  unsigned short* Wqb = p; p += nWQ;
  unsigned short* Wkb = p; p += nWK;
  unsigned short* Wvb = p; p += nWK;
  unsigned short* Wob = p; p += nWQ;
  unsigned short* Qb  = p; p += nHS;
  unsigned short* Kb  = p; p += nKV;
  unsigned short* Vb  = p; p += nKV;
  unsigned short* Ob  = p; p += nHS;

  cvt_f32_to_bf16<<<4096, 256, 0, stream>>>(hs, hsb, nHS);
  cvt_f32_to_bf16<<<4096, 256, 0, stream>>>(Wq, Wqb, nWQ);
  cvt_f32_to_bf16<<<1024, 256, 0, stream>>>(Wk, Wkb, nWK);
  cvt_f32_to_bf16<<<1024, 256, 0, stream>>>(Wv, Wvb, nWK);
  cvt_f32_to_bf16<<<4096, 256, 0, stream>>>(Wo, Wob, nWQ);

  const int M = BATCH * SEQ;  // 4096
  gemm_bias_wmma<true ><<<dim3(HIDN / 128, M / 128), 256, 0, stream>>>(
      hsb, Wqb, bq, Qb, M, HIDN, HIDN);
  gemm_bias_wmma<true ><<<dim3(HD / 128,   M / 128), 256, 0, stream>>>(
      hsb, Wkb, bk, Kb, M, HD, HIDN);
  gemm_bias_wmma<true ><<<dim3(HD / 128,   M / 128), 256, 0, stream>>>(
      hsb, Wvb, bv, Vb, M, HD, HIDN);

  const size_t smem = (size_t)(128 + 128 + 8 * 16) * 136 * sizeof(unsigned short); // 104448 B (< 320KB WGP LDS)
  gqa_flash_wmma<<<dim3(SEQ / 128, BATCH * NH), 256, smem, stream>>>(
      Qb, Kb, Vb, pmask, Ob);

  gemm_bias_wmma<false><<<dim3(HIDN / 128, M / 128), 256, 0, stream>>>(
      Ob, Wob, bo, d_out, M, HIDN, HIDN);
}
